// MixtureOfExperts_57019985822196
// MI455X (gfx1250) — matrix operand order repaired and checked by reading, not compiled, not run on hardware
//
#include <hip/hip_runtime.h>
#include <hip/hip_bf16.h>

#define T_TOK 4096
#define H_DIM 1024
#define F_DIM 4096
#define E_NUM 8
#define M_TILE 32

typedef unsigned short u16;
typedef __attribute__((ext_vector_type(16))) __bf16 v16bf;
typedef __attribute__((ext_vector_type(8)))  float  v8f;
typedef __attribute__((ext_vector_type(4))) unsigned int v4u;
typedef __attribute__((ext_vector_type(8)))  int v8i;
typedef __attribute__((ext_vector_type(4)))  int v4i;

union FragB { v16bf bf; uint4 q[2]; };

// ---- LDS layout (bytes). All offsets 16B-aligned; strides padded vs banks.
#define A_STRIDE   1032                 // u16 (1024 + 8 pad)
#define LDS_A_OFF  0                    // 32*1032*2 = 66048
#define WT_STRIDE  40                   // u16 (32 + 8 pad)
#define LDS_WI0    66048                // 64*40*2 = 5120
#define LDS_WG0    71168
#define LDS_WI1    76288
#define LDS_WG1    81408
#define H_STRIDE   72                   // u16 (64 + 8 pad)
#define LDS_H      86528                // 32*72*2 = 4608
#define WOT_STRIDE 72                   // u16 (64 + 8 pad)
#define LDS_WOT    91136                // 64*72*2 = 9216
#define LDS_WOS0   100352               // TDM staging: 64*64*4 = 16384
#define LDS_WOS1   116736               // second buffer (ends 133120)
#define R_STRIDE   1032                 // float; overlays [0,132096) at finalize
#define LDS_TOK    133120               // 32 ints
#define LDS_W      133248               // 32 floats
#define LDS_BYTES  133376

// Native bf16 converts (v_cvt_pk_bf16_f32 on gfx1250, RNE).
__device__ __forceinline__ unsigned pk_bf16(float a, float b) {
  union { __bf16 h; u16 u; } x, y;
  x.h = (__bf16)a; y.h = (__bf16)b;
  return (unsigned)x.u | ((unsigned)y.u << 16);
}
__device__ __forceinline__ u16 bf16_1(float a) {
  union { __bf16 h; u16 u; } x;
  x.h = (__bf16)a;
  return x.u;
}

// ---- Tensor Data Mover: 2-D fp32 tile (tile_d1 rows x tile_d0 cols) -> LDS.
// D# packing per CDNA5 ISA 8.3/8.4: count=1, type=2 (image), data_size=4B.
__device__ __forceinline__ void tdm_load_tile_f32(
    unsigned ldsOff, const float* g, unsigned tile_d0, unsigned tile_d1,
    unsigned tensor_d0, unsigned tensor_d1, unsigned stride0) {
  unsigned long long ga = (unsigned long long)g;
  v4u g0;
  g0[0] = 1u;                                          // count=1, user desc
  g0[1] = ldsOff;                                      // lds_addr (bytes)
  g0[2] = (unsigned)ga;                                // global_addr[31:0]
  g0[3] = (unsigned)((ga >> 32) & 0x01ffffffu) | 0x80000000u;  // addr[56:32]|type=2
  v8i g1;
  g1[0] = 0x20000;                                     // data_size=2 -> 4 bytes
  g1[1] = (int)((tensor_d0 & 0xffffu) << 16);          // tensor_dim0[15:0]
  g1[2] = (int)((tensor_d0 >> 16) | ((tensor_d1 & 0xffffu) << 16));
  g1[3] = (int)((tensor_d1 >> 16) | (tile_d0 << 16));  // tile_dim0
  g1[4] = (int)tile_d1;                                // tile_dim1 (tile_dim2=0)
  g1[5] = (int)stride0;                                // tensor_dim0_stride lo
  g1[6] = 0;
  g1[7] = 0;
  v4i z4 = {0, 0, 0, 0};
#if defined(__clang_major__) && __clang_major__ >= 23
  v8i z8 = {0, 0, 0, 0, 0, 0, 0, 0};
  __builtin_amdgcn_tensor_load_to_lds(g0, g1, z4, z4, z8, 0);
#else
  __builtin_amdgcn_tensor_load_to_lds(g0, g1, z4, z4, 0);
#endif
}

// ---------------------------------------------------------------- router ----
__global__ void moe_router(const float* __restrict__ x, const float* __restrict__ Wr,
                           const float* __restrict__ br, float* __restrict__ wdense) {
  int t = blockIdx.x * blockDim.x + threadIdx.x;
  if (t >= T_TOK) return;
  float lg[E_NUM];
#pragma unroll
  for (int e = 0; e < E_NUM; ++e) lg[e] = br[e];
  const float* xr = x + (size_t)t * H_DIM;
  for (int h = 0; h < H_DIM; ++h) {
    float xv = xr[h];
#pragma unroll
    for (int e = 0; e < E_NUM; ++e) lg[e] += xv * Wr[h * E_NUM + e];
  }
  float mx = lg[0];
#pragma unroll
  for (int e = 1; e < E_NUM; ++e) mx = fmaxf(mx, lg[e]);
  float p[E_NUM]; float s = 0.f;
#pragma unroll
  for (int e = 0; e < E_NUM; ++e) { p[e] = __expf(lg[e] - mx); s += p[e]; }
  float inv = 1.f / s;
#pragma unroll
  for (int e = 0; e < E_NUM; ++e) p[e] *= inv;
  int i0 = 0;
#pragma unroll
  for (int e = 1; e < E_NUM; ++e) if (p[e] > p[i0]) i0 = e;
  int i1 = (i0 == 0) ? 1 : 0;
#pragma unroll
  for (int e = 0; e < E_NUM; ++e) if (e != i0 && p[e] > p[i1]) i1 = e;
  float s2 = p[i0] + p[i1];
#pragma unroll
  for (int e = 0; e < E_NUM; ++e) wdense[t * E_NUM + e] = 0.f;
  wdense[t * E_NUM + i0] = p[i0] / s2;
  wdense[t * E_NUM + i1] = p[i1] / s2;
}

// --------------------------------------------- deterministic list builder ----
__global__ void moe_build_lists(const float* __restrict__ wdense,
                                int* __restrict__ tlist, int* __restrict__ counts) {
  int e = blockIdx.x;
  int tid = threadIdx.x;
  __shared__ int sc[256];
  __shared__ int base_s;
  if (tid == 0) base_s = 0;
  __syncthreads();
  for (int t0 = 0; t0 < T_TOK; t0 += 256) {
    int t = t0 + tid;
    int f = (wdense[t * E_NUM + e] > 0.f) ? 1 : 0;
    sc[tid] = f;
    __syncthreads();
    for (int off = 1; off < 256; off <<= 1) {
      int v = (tid >= off) ? sc[tid - off] : 0;
      __syncthreads();
      sc[tid] += v;
      __syncthreads();
    }
    int b = base_s;
    if (f) tlist[e * T_TOK + b + sc[tid] - 1] = t;
    __syncthreads();
    if (tid == 0) base_s = b + sc[255];
    __syncthreads();
  }
  if (tid == 0) counts[e] = base_s;
}

// ------------------------------------------------------------------ zero ----
__global__ void zero_out(float* __restrict__ out, int n) {
  int i = blockIdx.x * blockDim.x + threadIdx.x;
  int stride = gridDim.x * blockDim.x;
  for (; i < n; i += stride) out[i] = 0.f;
}

// --------------------------------------------------- grouped-GEMM expert ----
__global__ void __launch_bounds__(256)
moe_ffn(const float* __restrict__ x,
        const float* __restrict__ Wi, const float* __restrict__ bi,
        const float* __restrict__ Wg, const float* __restrict__ bg,
        const float* __restrict__ Wo, const float* __restrict__ bo,
        const float* __restrict__ gamma, const float* __restrict__ beta,
        const float* __restrict__ wdense, const int* __restrict__ tlist,
        const int* __restrict__ counts, float* __restrict__ out) {
  extern __shared__ char smem[];
  u16*  As   = (u16*)(smem + LDS_A_OFF);
  u16*  Hs   = (u16*)(smem + LDS_H);
  u16*  WoT  = (u16*)(smem + LDS_WOT);
  int*  rowtok = (int*)(smem + LDS_TOK);
  float* roww  = (float*)(smem + LDS_W);

  const int e = blockIdx.y;
  const int cnt = counts[e];
  const int row0 = blockIdx.x * M_TILE;
  if (row0 >= cnt) return;

  const int tid  = threadIdx.x;
  const int wid  = tid >> 5;
  const int lane = tid & 31;
  const int l16  = lane & 15;
  const int lhi  = lane >> 4;              // 0 or 1
  const int mh   = wid & 1;                // M half
  const int nq   = wid >> 1;               // 0..3
  const int mrow = 16 * mh + l16;          // A-fragment row for this lane

  const float* Wie = Wi + (size_t)e * H_DIM * F_DIM;
  const float* Wge = Wg + (size_t)e * H_DIM * F_DIM;
  const float* Woe = Wo + (size_t)e * F_DIM * H_DIM;
  const float* bie = bi + (size_t)e * F_DIM;
  const float* bge = bg + (size_t)e * F_DIM;
  const float* boe = bo + (size_t)e * H_DIM;

  if (tid < M_TILE) {
    int idx = row0 + tid;
    int tok = 0; float w = 0.f;
    if (idx < cnt) { tok = tlist[e * T_TOK + idx]; w = wdense[tok * E_NUM + e]; }
    rowtok[tid] = tok; roww[tid] = w;
  }
  __syncthreads();

  // Stage A tile: 32 token rows of x, fp32 -> packed bf16 LDS.
  for (int i = tid; i < M_TILE * (H_DIM / 4); i += 256) {
    int r  = i >> 8;
    int c4 = i & 255;
    float4 v = ((const float4*)(x + (size_t)rowtok[r] * H_DIM))[c4];
    unsigned* d32 = (unsigned*)(As + r * A_STRIDE + c4 * 4);
    d32[0] = pk_bf16(v.x, v.y);
    d32[1] = pk_bf16(v.z, v.w);
  }

  // Per-thread staging indices: k-pair (kp, kp+1), 4 columns starting at n4.
  const int kp = (tid >> 4) * 2;           // 0,2,...,30
  const int n4 = (tid & 15) * 4;           // 0..60

  v8f acc[16];
#pragma unroll
  for (int i = 0; i < 16; ++i)
#pragma unroll
    for (int v = 0; v < 8; ++v) acc[i][v] = 0.f;

  for (int fc = 0; fc < F_DIM; fc += 64) {
    // Arm TDM: first two Wo tiles of this F-chunk DMA while GEMM1 runs.
    if (wid == 0) {
      tdm_load_tile_f32(LDS_WOS0, Woe + (size_t)fc * H_DIM + 0 * 64,
                        64, 64, H_DIM, F_DIM, H_DIM);
      tdm_load_tile_f32(LDS_WOS1, Woe + (size_t)fc * H_DIM + 1 * 64,
                        64, 64, H_DIM, F_DIM, H_DIM);
    }

    // ---- GEMM1: [32,H] @ Wi/Wg[H, fc:fc+64], software-pipelined staging ----
    v8f ci, cg;
#pragma unroll
    for (int v = 0; v < 8; ++v) { ci[v] = 0.f; cg[v] = 0.f; }

    float4 ri0, ri1, rg0, rg1;               // rows kp and kp+1 of next tile
    {
      size_t ga = (size_t)kp * F_DIM + fc + n4;
      ri0 = *(const float4*)(Wie + ga); ri1 = *(const float4*)(Wie + ga + F_DIM);
      rg0 = *(const float4*)(Wge + ga); rg1 = *(const float4*)(Wge + ga + F_DIM);
      u16* wi = (u16*)(smem + LDS_WI0);
      u16* wg = (u16*)(smem + LDS_WG0);
      *(unsigned*)(wi + (n4+0)*WT_STRIDE + kp) = pk_bf16(ri0.x, ri1.x);
      *(unsigned*)(wi + (n4+1)*WT_STRIDE + kp) = pk_bf16(ri0.y, ri1.y);
      *(unsigned*)(wi + (n4+2)*WT_STRIDE + kp) = pk_bf16(ri0.z, ri1.z);
      *(unsigned*)(wi + (n4+3)*WT_STRIDE + kp) = pk_bf16(ri0.w, ri1.w);
      *(unsigned*)(wg + (n4+0)*WT_STRIDE + kp) = pk_bf16(rg0.x, rg1.x);
      *(unsigned*)(wg + (n4+1)*WT_STRIDE + kp) = pk_bf16(rg0.y, rg1.y);
      *(unsigned*)(wg + (n4+2)*WT_STRIDE + kp) = pk_bf16(rg0.z, rg1.z);
      *(unsigned*)(wg + (n4+3)*WT_STRIDE + kp) = pk_bf16(rg0.w, rg1.w);
    }
    int cur = 0;
    for (int k0 = 0; k0 < H_DIM; k0 += 32) {
      const bool more = (k0 + 32) < H_DIM;
      if (more) {
        size_t ga = (size_t)(k0 + 32 + kp) * F_DIM + fc + n4;
        ri0 = *(const float4*)(Wie + ga); ri1 = *(const float4*)(Wie + ga + F_DIM);
        rg0 = *(const float4*)(Wge + ga); rg1 = *(const float4*)(Wge + ga + F_DIM);
        if (k0 + 64 < H_DIM) {             // L2 prefetch two steps ahead
          __builtin_prefetch(Wie + (size_t)(k0 + 64 + kp) * F_DIM + fc + n4, 0, 1);
          __builtin_prefetch(Wge + (size_t)(k0 + 64 + kp) * F_DIM + fc + n4, 0, 1);
        }
      }
      __syncthreads();
      FragB af, bwi, bwg;
      const u16* ap = As + mrow * A_STRIDE + k0 + lhi * 8;
      af.q[0] = *(const uint4*)(ap);
      af.q[1] = *(const uint4*)(ap + 16);
      const u16* bp = (const u16*)(smem + (cur ? LDS_WI1 : LDS_WI0))
                      + (16 * nq + l16) * WT_STRIDE + lhi * 8;
      bwi.q[0] = *(const uint4*)(bp);
      bwi.q[1] = *(const uint4*)(bp + 16);
      const u16* gp = (const u16*)(smem + (cur ? LDS_WG1 : LDS_WG0))
                      + (16 * nq + l16) * WT_STRIDE + lhi * 8;
      bwg.q[0] = *(const uint4*)(gp);
      bwg.q[1] = *(const uint4*)(gp + 16);
      ci = __builtin_amdgcn_wmma_f32_16x16x32_bf16(false, af.bf, false, bwi.bf,
                                                   (short)0, ci, false, false);
      cg = __builtin_amdgcn_wmma_f32_16x16x32_bf16(false, af.bf, false, bwg.bf,
                                                   (short)0, cg, false, false);
      if (more) {
        u16* wi = (u16*)(smem + (cur ? LDS_WI0 : LDS_WI1));   // cur ^ 1
        u16* wg = (u16*)(smem + (cur ? LDS_WG0 : LDS_WG1));
        *(unsigned*)(wi + (n4+0)*WT_STRIDE + kp) = pk_bf16(ri0.x, ri1.x);
        *(unsigned*)(wi + (n4+1)*WT_STRIDE + kp) = pk_bf16(ri0.y, ri1.y);
        *(unsigned*)(wi + (n4+2)*WT_STRIDE + kp) = pk_bf16(ri0.z, ri1.z);
        *(unsigned*)(wi + (n4+3)*WT_STRIDE + kp) = pk_bf16(ri0.w, ri1.w);
        *(unsigned*)(wg + (n4+0)*WT_STRIDE + kp) = pk_bf16(rg0.x, rg1.x);
        *(unsigned*)(wg + (n4+1)*WT_STRIDE + kp) = pk_bf16(rg0.y, rg1.y);
        *(unsigned*)(wg + (n4+2)*WT_STRIDE + kp) = pk_bf16(rg0.z, rg1.z);
        *(unsigned*)(wg + (n4+3)*WT_STRIDE + kp) = pk_bf16(rg0.w, rg1.w);
      }
      cur ^= 1;
    }

    // ---- fused bias + exact GELU * sigmoid -> bf16 h tile ----
    {
      int nc = fc + 16 * nq + l16;
      float biv = bie[nc], bgv = bge[nc];
#pragma unroll
      for (int v = 0; v < 8; ++v) {
        float hv = ci[v] + biv;
        float gv = cg[v] + bgv;
        float ge = 0.5f * hv * (1.f + erff(hv * 0.70710678118654752f));
        float sg = 1.f / (1.f + __expf(-gv));
        int m = 16 * mh + (lhi << 3) + v;
        Hs[m * H_STRIDE + 16 * nq + l16] = bf16_1(ge * sg);
      }
    }

    // ---- GEMM2: acc += h[32,64] @ Wo[fc:fc+64, :], TDM double-buffered ----
#pragma unroll
    for (int ng = 0; ng < 16; ++ng) {
      if (wid == 0) {
        if (ng == 15) __builtin_amdgcn_s_wait_tensorcnt(0);   // literal imm
        else          __builtin_amdgcn_s_wait_tensorcnt(1);   // literal imm
      }
      __syncthreads();                      // publish TDM tile; h write/read order
      {
        const float* st = (const float*)(smem + ((ng & 1) ? LDS_WOS1 : LDS_WOS0));
        for (int i = tid; i < 512; i += 256) {   // 64x64 fp32 -> WoT[n][k] packed
          int kp2 = (i >> 4) * 2, m4 = (i & 15) * 4;
          float4 w0 = *(const float4*)(st + kp2 * 64 + m4);
          float4 w1 = *(const float4*)(st + (kp2 + 1) * 64 + m4);
          *(unsigned*)(WoT + (m4+0)*WOT_STRIDE + kp2) = pk_bf16(w0.x, w1.x);
          *(unsigned*)(WoT + (m4+1)*WOT_STRIDE + kp2) = pk_bf16(w0.y, w1.y);
          *(unsigned*)(WoT + (m4+2)*WOT_STRIDE + kp2) = pk_bf16(w0.z, w1.z);
          *(unsigned*)(WoT + (m4+3)*WOT_STRIDE + kp2) = pk_bf16(w0.w, w1.w);
        }
      }
      __syncthreads();
      if (wid == 0 && ng + 2 < 16)          // re-arm consumed staging buffer
        tdm_load_tile_f32((ng & 1) ? LDS_WOS1 : LDS_WOS0,
                          Woe + (size_t)fc * H_DIM + (ng + 2) * 64,
                          64, 64, H_DIM, F_DIM, H_DIM);
      v8f c = acc[ng];
#pragma unroll
      for (int kk = 0; kk < 64; kk += 32) {
        FragB ah, bw;
        const u16* hp = Hs + mrow * H_STRIDE + kk + lhi * 8;
        ah.q[0] = *(const uint4*)(hp);
        ah.q[1] = *(const uint4*)(hp + 16);
        const u16* wp = WoT + (16 * nq + l16) * WOT_STRIDE + kk + lhi * 8;
        bw.q[0] = *(const uint4*)(wp);
        bw.q[1] = *(const uint4*)(wp + 16);
        c = __builtin_amdgcn_wmma_f32_16x16x32_bf16(false, ah.bf, false, bw.bf,
                                                    (short)0, c, false, false);
      }
      acc[ng] = c;
    }
  }

  // ---- residual + bo into LDS (overlays dead regions), then LayerNorm ----
  __syncthreads();
  float* Rs = (float*)(smem);
#pragma unroll
  for (int ng = 0; ng < 16; ++ng) {
    int n = 64 * ng + 16 * nq + l16;
    float bov = boe[n];
#pragma unroll
    for (int v = 0; v < 8; ++v) {
      int m = 16 * mh + (lhi << 3) + v;
      Rs[m * R_STRIDE + n] = acc[ng][v] + bov + x[(size_t)rowtok[m] * H_DIM + n];
    }
  }
  __syncthreads();

  const float* ge_ = gamma + (size_t)e * H_DIM;
  const float* be_ = beta  + (size_t)e * H_DIM;
#pragma unroll
  for (int jr = 0; jr < 4; ++jr) {
    int m = wid * 4 + jr;
    const float* rr = Rs + m * R_STRIDE;
    float s = 0.f;
#pragma unroll
    for (int i = 0; i < 32; ++i) s += rr[lane + 32 * i];
#pragma unroll
    for (int off = 16; off > 0; off >>= 1) s += __shfl_xor(s, off, 32);
    float mu = s * (1.f / (float)H_DIM);
    float s2 = 0.f;
#pragma unroll
    for (int i = 0; i < 32; ++i) { float d = rr[lane + 32 * i] - mu; s2 += d * d; }
#pragma unroll
    for (int off = 16; off > 0; off >>= 1) s2 += __shfl_xor(s2, off, 32);
    float rstd = rsqrtf(s2 * (1.f / (float)H_DIM) + 1e-12f);
    float wgt = roww[m];
    if (wgt != 0.f) {
      int tok = rowtok[m];
#pragma unroll
      for (int i = 0; i < 32; ++i) {
        int n = lane + 32 * i;
        float val = (rr[n] - mu) * rstd * ge_[n] + be_[n];
        atomicAdd(out + (size_t)tok * H_DIM + n, wgt * val);
      }
    }
  }
}

// ---------------------------------------------------------------- launch ----
extern "C" void kernel_launch(void* const* d_in, const int* in_sizes, int n_in,
                              void* d_out, int out_size, void* d_ws, size_t ws_size,
                              hipStream_t stream) {
  (void)in_sizes; (void)n_in; (void)ws_size;
  const float* x     = (const float*)d_in[0];
  const float* Wr    = (const float*)d_in[1];
  const float* br    = (const float*)d_in[2];
  const float* Wi    = (const float*)d_in[3];
  const float* bi    = (const float*)d_in[4];
  const float* Wg    = (const float*)d_in[5];
  const float* bg    = (const float*)d_in[6];
  const float* Wo    = (const float*)d_in[7];
  const float* bo    = (const float*)d_in[8];
  const float* gamma = (const float*)d_in[9];
  const float* beta  = (const float*)d_in[10];
  float* out = (float*)d_out;

  char* ws = (char*)d_ws;
  float* wdense = (float*)ws;                                   // T*E floats
  int*   counts = (int*)(ws + (size_t)T_TOK * E_NUM * 4);       // E ints (+pad)
  int*   tlist  = (int*)(ws + (size_t)T_TOK * E_NUM * 4 + 64);  // E*T ints

  moe_router<<<T_TOK / 128, 128, 0, stream>>>(x, Wr, br, wdense);
  moe_build_lists<<<E_NUM, 256, 0, stream>>>(wdense, tlist, counts);
  zero_out<<<256, 256, 0, stream>>>(out, out_size);
  dim3 grid(T_TOK / M_TILE, E_NUM);
  moe_ffn<<<grid, 256, LDS_BYTES, stream>>>(x, Wi, bi, Wg, bg, Wo, bo,
                                            gamma, beta, wdense, tlist, counts, out);
}
